// PointNetSetAbstraction_68745246539910
// MI455X (gfx1250) — matrix operand order repaired and checked by reading, not compiled, not run on hardware
//
#include <hip/hip_runtime.h>
#include <hip/hip_bf16.h>

typedef __attribute__((ext_vector_type(16))) _Float16 v16h;
typedef __attribute__((ext_vector_type(8)))  float    v8f;
typedef __attribute__((ext_vector_type(4)))  float    vf4;

#define EPS_BN 1e-5f

__device__ __forceinline__ v8f wmma_f32_f16(v16h a, v16h b, v8f c) {
  // D = A(16x32 f16) x B(32x16 f16) + C(16x16 f32)
  return __builtin_amdgcn_wmma_f32_16x16x32_f16(false, a, false, b, (short)0, c,
                                                false, false);
}

// ---------------- kernel: gather new_bxyz = point_bxyz[sample_idx] ----------------
__global__ void k_gather_bxyz(const float* __restrict__ bxyz,
                              const int* __restrict__ sidx,
                              float* __restrict__ out, int M4) {
  int i = blockIdx.x * blockDim.x + threadIdx.x;
  if (i < M4) {
    int m = i >> 2, c = i & 3;
    out[i] = bxyz[(size_t)sidx[m] * 4 + c];
  }
}

// ---- fused: edge_in = [rel_xyz | feat_gather]; h = relu(edge_in@W0+b0); segmax ----
__global__ void __launch_bounds__(256)
k_edge_gemm_segmax(const float* __restrict__ bxyz,
                   const float* __restrict__ feat,
                   const int* __restrict__ sidx,
                   const int* __restrict__ e_point,
                   const int* __restrict__ e_new,
                   const float* __restrict__ W0,
                   const float* __restrict__ b0,
                   unsigned* __restrict__ agg,   // [M,64] fp32 bit-patterns, zeroed
                   int num_tiles) {
  const int lane = threadIdx.x & 31;
  const int hh   = lane >> 4;       // half-wave select
  const int lr   = lane & 15;
  int wave  = blockIdx.x * (blockDim.x >> 5) + (threadIdx.x >> 5);
  int nwave = gridDim.x * (blockDim.x >> 5);

  // Keep all of W0 (64x64) resident: 4 N-tiles x 2 K-tiles of B fragments.
  v16h bf[4][2];
#pragma unroll
  for (int nt = 0; nt < 4; ++nt)
#pragma unroll
    for (int kt = 0; kt < 2; ++kt) {
      const int n  = nt * 16 + lr;
      const int kb = kt * 32 + hh * 8;      // B: K = kb + j + 8*(j>>3) (ISA table order)
      v16h b;
#pragma unroll
      for (int j = 0; j < 16; ++j) b[j] = (_Float16)W0[(size_t)(kb + j + ((j >> 3) << 3)) * 64 + n];
      bf[nt][kt] = b;
    }
  float biasv[4];
#pragma unroll
  for (int nt = 0; nt < 4; ++nt) biasv[nt] = b0[nt * 16 + lr];

  for (int t = wave; t < num_tiles; t += nwave) {
    const int e0 = t * 16;
    const int e  = e0 + lr;
    const int p  = e_point[e];
    const int sg = e_new[e];
    const int s  = sidx[sg];

    // A fragments: ISA layout -> element j holds K = kt*32 + hh*8 + j + 8*(j>>3)
    v16h af[2];
#pragma unroll
    for (int kt = 0; kt < 2; ++kt) {
      v16h a;
#pragma unroll
      for (int j = 0; j < 16; ++j) {
        const int k = kt * 32 + hh * 8 + j + ((j >> 3) << 3);
        float v;
        if (k < 3) v = bxyz[(size_t)p * 4 + 1 + k] - bxyz[(size_t)s * 4 + 1 + k];
        else       v = feat[(size_t)p * 61 + (k - 3)];
        a[j] = (_Float16)v;
      }
      af[kt] = a;
    }

    v8f c[4];
#pragma unroll
    for (int nt = 0; nt < 4; ++nt) {
      v8f acc = {biasv[nt], biasv[nt], biasv[nt], biasv[nt],
                 biasv[nt], biasv[nt], biasv[nt], biasv[nt]};
      acc = wmma_f32_f16(af[0], bf[nt][0], acc);
      acc = wmma_f32_f16(af[1], bf[nt][1], acc);
      c[nt] = acc;
    }

    const int seg_first = e_new[e0];
    const int seg_last  = e_new[e0 + 15];
    if (seg_first == seg_last) {
      // common case (mean degree 32, e_new sorted): whole tile -> one segment.
#pragma unroll
      for (int nt = 0; nt < 4; ++nt) {
        float v = 0.0f;                         // 0-start folds ReLU
#pragma unroll
        for (int r = 0; r < 8; ++r) v = fmaxf(v, c[nt][r]);
        v = fmaxf(v, __shfl_xor(v, 16, 32));    // combine the two row halves
        if (lane < 16)
          atomicMax(&agg[(size_t)seg_first * 64 + nt * 16 + lane],
                    __float_as_uint(v));
      }
    } else {
      int segs[8];
#pragma unroll
      for (int r = 0; r < 8; ++r) segs[r] = e_new[e0 + r + hh * 8];
#pragma unroll
      for (int nt = 0; nt < 4; ++nt)
#pragma unroll
        for (int r = 0; r < 8; ++r) {
          const float v = fmaxf(c[nt][r], 0.0f);
          atomicMax(&agg[(size_t)segs[r] * 64 + nt * 16 + lr],
                    __float_as_uint(v));
        }
    }
  }
}

// ---------------- generic WMMA GEMM + bias: Y[M,N] = X[M,K] @ W[K,N] + b ----------------
template <int K, int N>
__global__ void __launch_bounds__(256)
k_gemm_bias(const float* __restrict__ X, const float* __restrict__ W,
            const float* __restrict__ bias, float* __restrict__ Y, int Mrows) {
  const int lane = threadIdx.x & 31;
  const int hh = lane >> 4, lr = lane & 15;
  int wave  = blockIdx.x * (blockDim.x >> 5) + (threadIdx.x >> 5);
  int nwave = gridDim.x * (blockDim.x >> 5);
  const int ntn   = N / 16;
  const int total = (Mrows / 16) * ntn;

  for (int t = wave; t < total; t += nwave) {
    const int mt = t / ntn, nt = t % ntn;
    const int n = nt * 16 + lr;
    const int m = mt * 16 + lr;
    const float bv = bias[n];
    v8f acc = {bv, bv, bv, bv, bv, bv, bv, bv};
#pragma unroll
    for (int kt = 0; kt < K / 32; ++kt) {
      // A: two contiguous 8-float chunks per lane (vectorized b128 loads)
      const float* ap = X + (size_t)m * K + kt * 32 + hh * 8;
      const vf4 a0 = ((const vf4*)ap)[0];
      const vf4 a1 = ((const vf4*)ap)[1];
      const vf4 a2 = ((const vf4*)(ap + 16))[0];
      const vf4 a3 = ((const vf4*)(ap + 16))[1];
      v16h a;
#pragma unroll
      for (int j = 0; j < 4; ++j) {
        a[j]      = (_Float16)a0[j];
        a[4 + j]  = (_Float16)a1[j];
        a[8 + j]  = (_Float16)a2[j];
        a[12 + j] = (_Float16)a3[j];
      }
      const int kb = kt * 32 + hh * 8;
      v16h b;
#pragma unroll
      for (int j = 0; j < 16; ++j) b[j] = (_Float16)W[(size_t)(kb + j + ((j >> 3) << 3)) * N + n];
      acc = wmma_f32_f16(a, b, acc);
    }
#pragma unroll
    for (int r = 0; r < 8; ++r)
      Y[(size_t)(mt * 16 + r + hh * 8) * N + nt * 16 + lr] = acc[r];
  }
}

// ---------------- per-column sum / sum-of-squares (blockDim.x == N) ----------------
__global__ void k_colstats(const float* __restrict__ Y, float* __restrict__ sum,
                           float* __restrict__ sq, int Mrows, int N) {
  const int c = threadIdx.x;
  const long r0 = (long)Mrows * blockIdx.x / gridDim.x;
  const long r1 = (long)Mrows * (blockIdx.x + 1) / gridDim.x;
  float s = 0.0f, q = 0.0f;
  for (long r = r0; r < r1; ++r) {
    const float v = Y[(size_t)r * N + c];
    s += v;
    q = fmaf(v, v, q);
  }
  atomicAdd(&sum[c], s);
  atomicAdd(&sq[c], q);
}

__global__ void k_bn_finalize(const float* __restrict__ sum, const float* __restrict__ sq,
                              const float* __restrict__ g, const float* __restrict__ beta,
                              float* __restrict__ scale, float* __restrict__ shift,
                              float invM, int N) {
  const int c = threadIdx.x;
  if (c < N) {
    const float mu  = sum[c] * invM;
    const float var = sq[c] * invM - mu * mu;
    const float sc  = g[c] * rsqrtf(var + EPS_BN);
    scale[c] = sc;
    shift[c] = fmaf(-mu, sc, beta[c]);
  }
}

// in-place y = relu(y*scale[c] + shift[c]);  N is a power of two
__global__ void k_bn_relu(float* __restrict__ Y, const float* __restrict__ scale,
                          const float* __restrict__ shift, size_t total, int Nmask) {
  size_t i = (size_t)blockIdx.x * blockDim.x + threadIdx.x;
  const size_t stride = (size_t)gridDim.x * blockDim.x;
  for (; i < total; i += stride) {
    const int c = (int)(i & (size_t)Nmask);
    Y[i] = fmaxf(fmaf(Y[i], scale[c], shift[c]), 0.0f);
  }
}

extern "C" void kernel_launch(void* const* d_in, const int* in_sizes, int n_in,
                              void* d_out, int out_size, void* d_ws, size_t ws_size,
                              hipStream_t stream) {
  const float* bxyz  = (const float*)d_in[0];
  const float* feat  = (const float*)d_in[1];
  const int*   sidx  = (const int*)d_in[2];
  const int*   e_pt  = (const int*)d_in[3];
  const int*   e_new = (const int*)d_in[4];
  const float* W0 = (const float*)d_in[5];
  const float* b0 = (const float*)d_in[6];
  const float* W1 = (const float*)d_in[7];
  const float* b1 = (const float*)d_in[8];
  const float* g1 = (const float*)d_in[9];
  const float* be1 = (const float*)d_in[10];
  const float* W2 = (const float*)d_in[11];
  const float* b2 = (const float*)d_in[12];
  const float* g2 = (const float*)d_in[13];
  const float* be2 = (const float*)d_in[14];

  const int Mm = in_sizes[2];          // 100000
  const int Ee = in_sizes[3];          // 3200000

  // workspace layout (floats)
  float* ws    = (float*)d_ws;
  float* agg   = ws;                               // [M,64]
  float* h1    = agg + (size_t)Mm * 64;            // [M,128]
  float* stats = h1 + (size_t)Mm * 128;            // 2048 floats
  float* sum1 = stats, *sq1 = stats + 256, *scale1 = stats + 512, *shift1 = stats + 768;
  float* sum2 = stats + 1024, *sq2 = stats + 1280, *scale2 = stats + 1536, *shift2 = stats + 1792;

  float* out_bxyz = (float*)d_out;                 // [M,4]
  float* out_x    = out_bxyz + (size_t)Mm * 4;     // [M,256]

  hipMemsetAsync(agg, 0, (size_t)Mm * 64 * sizeof(float), stream);
  hipMemsetAsync(stats, 0, 2048 * sizeof(float), stream);

  // new_bxyz gather
  k_gather_bxyz<<<(Mm * 4 + 255) / 256, 256, 0, stream>>>(bxyz, sidx, out_bxyz, Mm * 4);

  // fused edge GEMM + ReLU + segment-max  (E/16 = 200000 tiles)
  k_edge_gemm_segmax<<<2048, 256, 0, stream>>>(bxyz, feat, sidx, e_pt, e_new, W0, b0,
                                               (unsigned*)agg, Ee / 16);

  // layer 1: y1 = agg @ W1 + b1 ; BN stats ; BN+ReLU in place
  {
    const int tiles = (Mm / 16) * (128 / 16);
    k_gemm_bias<64, 128><<<(tiles + 7) / 8, 256, 0, stream>>>(agg, W1, b1, h1, Mm);
    k_colstats<<<256, 128, 0, stream>>>(h1, sum1, sq1, Mm, 128);
    k_bn_finalize<<<1, 128, 0, stream>>>(sum1, sq1, g1, be1, scale1, shift1, 1.0f / Mm, 128);
    k_bn_relu<<<4096, 256, 0, stream>>>(h1, scale1, shift1, (size_t)Mm * 128, 127);
  }

  // layer 2: y2 = x1 @ W2 + b2 straight into d_out ; BN stats ; BN+ReLU in place
  {
    const int tiles = (Mm / 16) * (256 / 16);
    k_gemm_bias<128, 256><<<(tiles + 7) / 8, 256, 0, stream>>>(h1, W2, b2, out_x, Mm);
    k_colstats<<<256, 256, 0, stream>>>(out_x, sum2, sq2, Mm, 256);
    k_bn_finalize<<<1, 256, 0, stream>>>(sum2, sq2, g2, be2, scale2, shift2, 1.0f / Mm, 256);
    k_bn_relu<<<8192, 256, 0, stream>>>(out_x, scale2, shift2, (size_t)Mm * 256, 255);
  }
}
